// _UnimolLayer_47579647705264
// MI455X (gfx1250) — matrix operand order repaired and checked by reading, not compiled, not run on hardware
//
#include <hip/hip_runtime.h>
#include <hip/hip_bf16.h>
#include <math.h>

// ---------------------------------------------------------------------------
// MI455X (gfx1250) implementation of a Unimol transformer layer.
// All GEMMs use v_wmma_f32_16x16x32_bf16 (bf16 in / f32 accumulate).
// Shapes: L=512 B=8 D=1024 H=16 hd=64 DFF=4096 -> M rows = L*B = 4096.
// ---------------------------------------------------------------------------

typedef __bf16 bf16_t;
typedef __attribute__((ext_vector_type(16))) __bf16 v16bf;
typedef __attribute__((ext_vector_type(8)))  __bf16 v8bf;
typedef __attribute__((ext_vector_type(8)))  float  v8f;

#define L_    512
#define B_    8
#define D_    1024
#define H_    16
#define HD_   64
#define DFF_  4096
#define ROWS  (L_ * B_)        // 4096
#define NHEAD (B_ * H_)        // 128

// ------------------------------ small kernels ------------------------------

__global__ void cast_f32_bf16(const float* __restrict__ in,
                              bf16_t* __restrict__ out, int n) {
  int i = blockIdx.x * blockDim.x + threadIdx.x;
  if (i < n) out[i] = (bf16_t)in[i];
}

// One 256-thread block per row of 1024 elements; writes bf16 normalized row.
__global__ void layernorm_bf16(const float* __restrict__ x,
                               const float* __restrict__ w,
                               const float* __restrict__ b,
                               bf16_t* __restrict__ out) {
  const int row = blockIdx.x;
  const int tid = threadIdx.x;
  const float* xr = x + (size_t)row * D_;
  float v[4], s = 0.f, s2 = 0.f;
#pragma unroll
  for (int i = 0; i < 4; ++i) {
    v[i] = xr[tid + i * 256];
    s  += v[i];
    s2 += v[i] * v[i];
  }
  __shared__ float red[256], red2[256];
  red[tid] = s; red2[tid] = s2;
  __syncthreads();
  for (int off = 128; off > 0; off >>= 1) {
    if (tid < off) { red[tid] += red[tid + off]; red2[tid] += red2[tid + off]; }
    __syncthreads();
  }
  const float mean = red[0] * (1.f / D_);
  const float var  = red2[0] * (1.f / D_) - mean * mean;
  const float rstd = rsqrtf(var + 1e-5f);
#pragma unroll
  for (int i = 0; i < 4; ++i) {
    int c = tid + i * 256;
    out[(size_t)row * D_ + c] = (bf16_t)(((v[i] - mean) * rstd) * w[c] + b[c]);
  }
}

// vt[(n*64+d)*512 + l] = V element (head n, key pos l, dim d) from qkv buffer.
__global__ void transpose_v(const bf16_t* __restrict__ qkv,
                            bf16_t* __restrict__ vt) {
  int idx = blockIdx.x * 256 + threadIdx.x;   // 128*64*512 total
  int l = idx & 511;
  int d = (idx >> 9) & 63;
  int n = idx >> 15;
  int b = n >> 4, h = n & 15;
  vt[idx] = qkv[((size_t)(l * 8 + b)) * 3072 + 2048 + h * 64 + d];
}

// ------------------------------- WMMA GEMM ---------------------------------
// C[M,N] = act(A[M,K] @ W[N,K]^T + bias) (+resid).  A,W bf16 row-major.
// Block = 128 threads = 4 waves; each wave owns a 16x64 output tile
// (4 f32 accumulators), K-loop step 32 -> A fragment reused across 4 WMMAs.
// Per-lane operand fragments (this code's contiguous-K order, same for A and B):
//   A 16x32: row = lane%16, 16 contiguous K at (lane/16)*16.
//   B 32x16: col = lane%16, 16 contiguous K at (lane/16)*16.
//   C 16x16: elem i -> (row = i + (lane/16)*8, col = lane%16).
__global__ void __launch_bounds__(128)
gemm_wmma(const bf16_t* __restrict__ A, const bf16_t* __restrict__ W,
          const float* __restrict__ bias, const float* __restrict__ resid,
          float* __restrict__ Cf, bf16_t* __restrict__ Cb,
          int M, int N, int K, int gelu) {
  const int lane = threadIdx.x & 31;
  const int wave = threadIdx.x >> 5;
  const int half = lane >> 4;        // 0 or 1
  const int l16  = lane & 15;
  const int m0 = blockIdx.y * 64 + wave * 16;
  const int n0 = blockIdx.x * 64;

  const bf16_t* arow  = A + (size_t)(m0 + l16) * K + half * 16;
  const bf16_t* wrow0 = W + (size_t)(n0 +  0 + l16) * K + half * 16;
  const bf16_t* wrow1 = W + (size_t)(n0 + 16 + l16) * K + half * 16;
  const bf16_t* wrow2 = W + (size_t)(n0 + 32 + l16) * K + half * 16;
  const bf16_t* wrow3 = W + (size_t)(n0 + 48 + l16) * K + half * 16;

  v8f acc[4] = {};
  for (int k0 = 0; k0 < K; k0 += 32) {
    v16bf a;
    ((v8bf*)&a)[0] = *(const v8bf*)(arow + k0);
    ((v8bf*)&a)[1] = *(const v8bf*)(arow + k0 + 8);
    v16bf b0, b1, b2, b3;
    ((v8bf*)&b0)[0] = *(const v8bf*)(wrow0 + k0);
    ((v8bf*)&b0)[1] = *(const v8bf*)(wrow0 + k0 + 8);
    ((v8bf*)&b1)[0] = *(const v8bf*)(wrow1 + k0);
    ((v8bf*)&b1)[1] = *(const v8bf*)(wrow1 + k0 + 8);
    ((v8bf*)&b2)[0] = *(const v8bf*)(wrow2 + k0);
    ((v8bf*)&b2)[1] = *(const v8bf*)(wrow2 + k0 + 8);
    ((v8bf*)&b3)[0] = *(const v8bf*)(wrow3 + k0);
    ((v8bf*)&b3)[1] = *(const v8bf*)(wrow3 + k0 + 8);
    acc[0] = __builtin_amdgcn_wmma_f32_16x16x32_bf16(false, a, false, b0,
                                                     (short)0, acc[0], false, false);
    acc[1] = __builtin_amdgcn_wmma_f32_16x16x32_bf16(false, a, false, b1,
                                                     (short)0, acc[1], false, false);
    acc[2] = __builtin_amdgcn_wmma_f32_16x16x32_bf16(false, a, false, b2,
                                                     (short)0, acc[2], false, false);
    acc[3] = __builtin_amdgcn_wmma_f32_16x16x32_bf16(false, a, false, b3,
                                                     (short)0, acc[3], false, false);
  }

  const int rbase = m0 + half * 8;
#pragma unroll
  for (int nt = 0; nt < 4; ++nt) {
    const int col = n0 + nt * 16 + l16;
    const float bia = bias ? bias[col] : 0.f;
#pragma unroll
    for (int i = 0; i < 8; ++i) {
      const int row = rbase + i;
      float v = acc[nt][i] + bia;
      if (gelu) v = 0.5f * v * (1.f + erff(v * 0.70710678118654752f));
      const size_t o = (size_t)row * N + col;
      if (resid) v += resid[o];
      if (Cf) Cf[o] = v;
      if (Cb) Cb[o] = (bf16_t)v;
    }
  }
}

// ------------------------------- attention ---------------------------------
// One wave per (head, 16-query-row strip).  Scores via WMMA (Q as A, K as B),
// scores*1/8 + edge stored to edge_out AND to a 16x512 f32 LDS strip;
// wave-local softmax; attn (bf16, LDS) @ V (pre-transposed) via WMMA.
#define SPITCH 520
__global__ void __launch_bounds__(32)
attn_kernel(const bf16_t* __restrict__ qkv, const bf16_t* __restrict__ vt,
            const float* __restrict__ edge, float* __restrict__ edge_out,
            bf16_t* __restrict__ ao) {
  __shared__ __align__(16) float  sbuf[16 * SPITCH];
  __shared__ __align__(16) bf16_t abuf[16 * 512];

  const int head  = blockIdx.x;          // 0..127  (= b*16 + h)
  const int strip = blockIdx.y;          // 0..31
  const int l0 = strip * 16;
  const int b = head >> 4, h = head & 15;
  const int lane = threadIdx.x & 31;
  const int half = lane >> 4;
  const int l16  = lane & 15;

  // Q tile [16 x 64] -> two A fragments (d 0..31 and 32..63)
  const bf16_t* qrow = qkv + ((size_t)((l0 + l16) * 8 + b)) * 3072 + h * 64;
  v16bf qa0, qa1;
  ((v8bf*)&qa0)[0] = *(const v8bf*)(qrow + half * 16);
  ((v8bf*)&qa0)[1] = *(const v8bf*)(qrow + half * 16 + 8);
  ((v8bf*)&qa1)[0] = *(const v8bf*)(qrow + 32 + half * 16);
  ((v8bf*)&qa1)[1] = *(const v8bf*)(qrow + 32 + half * 16 + 8);

  const float scale = 0.125f;            // 1/sqrt(64)
  for (int mt = 0; mt < 32; ++mt) {
    const int m0 = mt * 16;
    // K^T as B: col = key position (m0+l16), contiguous head-dim chunk per lane
    const bf16_t* krow =
        qkv + ((size_t)((m0 + l16) * 8 + b)) * 3072 + 1024 + h * 64 + half * 16;
    v16bf kb0, kb1;
    ((v8bf*)&kb0)[0] = *(const v8bf*)(krow);
    ((v8bf*)&kb0)[1] = *(const v8bf*)(krow + 8);
    ((v8bf*)&kb1)[0] = *(const v8bf*)(krow + 32);
    ((v8bf*)&kb1)[1] = *(const v8bf*)(krow + 40);
    v8f s = {};
    s = __builtin_amdgcn_wmma_f32_16x16x32_bf16(false, qa0, false, kb0,
                                                (short)0, s, false, false);
    s = __builtin_amdgcn_wmma_f32_16x16x32_bf16(false, qa1, false, kb1,
                                                (short)0, s, false, false);
    const int r0 = half * 8;
    const int cm = m0 + l16;
#pragma unroll
    for (int i = 0; i < 8; ++i) {
      const int r = r0 + i;
      const size_t e = ((size_t)head * 512 + (l0 + r)) * 512 + cm;
      const float v = s[i] * scale + edge[e];
      edge_out[e] = v;                   // edge_out = t + edge (pre-softmax)
      sbuf[r * SPITCH + cm] = v;
    }
  }
  __syncthreads();

  // row-wise softmax over 512 keys (shuffle reductions, wave32)
  for (int r = 0; r < 16; ++r) {
    float mx = -3.4e38f;
    for (int c = lane; c < 512; c += 32) mx = fmaxf(mx, sbuf[r * SPITCH + c]);
#pragma unroll
    for (int off = 16; off > 0; off >>= 1) mx = fmaxf(mx, __shfl_xor(mx, off, 32));
    float sum = 0.f;
    for (int c = lane; c < 512; c += 32) {
      const float e = __expf(sbuf[r * SPITCH + c] - mx);
      sbuf[r * SPITCH + c] = e;
      sum += e;
    }
#pragma unroll
    for (int off = 16; off > 0; off >>= 1) sum += __shfl_xor(sum, off, 32);
    const float inv = 1.f / sum;
    for (int c = lane; c < 512; c += 32)
      abuf[r * 512 + c] = (bf16_t)(sbuf[r * SPITCH + c] * inv);
  }
  __syncthreads();

  // attn[16,512] @ V[512,64] : A from LDS, B from pre-transposed vt
  v8f acc[4] = {};
  for (int kk = 0; kk < 16; ++kk) {
    const int k0 = kk * 32;
    v16bf a;
    ((v8bf*)&a)[0] = *(const v8bf*)(abuf + l16 * 512 + k0 + half * 16);
    ((v8bf*)&a)[1] = *(const v8bf*)(abuf + l16 * 512 + k0 + half * 16 + 8);
#pragma unroll
    for (int nt = 0; nt < 4; ++nt) {
      const bf16_t* vrow =
          vt + ((size_t)head * 64 + nt * 16 + l16) * 512 + k0 + half * 16;
      v16bf bb;
      ((v8bf*)&bb)[0] = *(const v8bf*)(vrow);
      ((v8bf*)&bb)[1] = *(const v8bf*)(vrow + 8);
      acc[nt] = __builtin_amdgcn_wmma_f32_16x16x32_bf16(false, a, false, bb,
                                                        (short)0, acc[nt],
                                                        false, false);
    }
  }

  // store ao in [L,B,D] layout -> directly feeds out-proj GEMM as A
  const int r0 = half * 8;
#pragma unroll
  for (int nt = 0; nt < 4; ++nt) {
#pragma unroll
    for (int i = 0; i < 8; ++i) {
      const int l   = l0 + r0 + i;
      const int col = h * 64 + nt * 16 + l16;
      ao[((size_t)l * 8 + b) * 1024 + col] = (bf16_t)acc[nt][i];
    }
  }
}

// ------------------------------- launcher ----------------------------------

extern "C" void kernel_launch(void* const* d_in, const int* in_sizes, int n_in,
                              void* d_out, int out_size, void* d_ws,
                              size_t ws_size, hipStream_t stream) {
  (void)in_sizes; (void)n_in; (void)out_size; (void)ws_size;
  const float* x     = (const float*)d_in[0];
  const float* edge  = (const float*)d_in[1];
  const float* in_w  = (const float*)d_in[2];
  const float* in_b  = (const float*)d_in[3];
  const float* out_w = (const float*)d_in[4];
  const float* out_b = (const float*)d_in[5];
  const float* l1_w  = (const float*)d_in[6];
  const float* l1_b  = (const float*)d_in[7];
  const float* l2_w  = (const float*)d_in[8];
  const float* l2_b  = (const float*)d_in[9];
  const float* n1_w  = (const float*)d_in[10];
  const float* n1_b  = (const float*)d_in[11];
  const float* n2_w  = (const float*)d_in[12];
  const float* n2_b  = (const float*)d_in[13];

  float* x2_out   = (float*)d_out;                    // [512,8,1024]
  float* edge_out = x2_out + (size_t)ROWS * D_;       // [128,512,512]

  char* ws = (char*)d_ws;
  size_t off = 0;
  bf16_t* qkvb = (bf16_t*)(ws + off);  off += (size_t)ROWS * 3 * D_ * 2;   // 24 MB
  bf16_t* vtb  = (bf16_t*)(ws + off);  off += (size_t)NHEAD * HD_ * L_ * 2;// 8 MB
  bf16_t* ff1  = qkvb;  // aliases qkv+vt region (dead after attention): 32 MB
  bf16_t* hb   = (bf16_t*)(ws + off);  off += (size_t)ROWS * D_ * 2;
  bf16_t* aob  = (bf16_t*)(ws + off);  off += (size_t)ROWS * D_ * 2;
  float*  x1   = (float*)(ws + off);   off += (size_t)ROWS * D_ * 4;
  bf16_t* wb_in  = (bf16_t*)(ws + off); off += (size_t)3 * D_ * D_ * 2;
  bf16_t* wb_out = (bf16_t*)(ws + off); off += (size_t)D_ * D_ * 2;
  bf16_t* wb_l1  = (bf16_t*)(ws + off); off += (size_t)DFF_ * D_ * 2;
  bf16_t* wb_l2  = (bf16_t*)(ws + off); off += (size_t)D_ * DFF_ * 2;

  const int th = 256;
  cast_f32_bf16<<<(3 * D_ * D_ + th - 1) / th, th, 0, stream>>>(in_w, wb_in, 3 * D_ * D_);
  cast_f32_bf16<<<(D_ * D_ + th - 1) / th, th, 0, stream>>>(out_w, wb_out, D_ * D_);
  cast_f32_bf16<<<(DFF_ * D_ + th - 1) / th, th, 0, stream>>>(l1_w, wb_l1, DFF_ * D_);
  cast_f32_bf16<<<(D_ * DFF_ + th - 1) / th, th, 0, stream>>>(l2_w, wb_l2, D_ * DFF_);

  // h = LN1(x)
  layernorm_bf16<<<ROWS, 256, 0, stream>>>(x, n1_w, n1_b, hb);
  // qkv = h @ in_proj_w^T + in_proj_b   [4096, 3072]
  gemm_wmma<<<dim3(3 * D_ / 64, ROWS / 64), 128, 0, stream>>>(
      hb, wb_in, in_b, nullptr, nullptr, qkvb, ROWS, 3 * D_, D_, 0);
  // V -> [head][64][512] for contiguous-K B fragments
  transpose_v<<<(NHEAD * HD_ * L_) / 256, 256, 0, stream>>>(qkvb, vtb);
  // attention + edge_out
  attn_kernel<<<dim3(NHEAD, L_ / 16), 32, 0, stream>>>(qkvb, vtb, edge,
                                                       edge_out, aob);
  // x1 = x + ao @ out_proj_w^T + out_proj_b
  gemm_wmma<<<dim3(D_ / 64, ROWS / 64), 128, 0, stream>>>(
      aob, wb_out, out_b, x, x1, nullptr, ROWS, D_, D_, 0);
  // h2 = LN2(x1)
  layernorm_bf16<<<ROWS, 256, 0, stream>>>(x1, n2_w, n2_b, hb);
  // ff1 = gelu(h2 @ lin1_w^T + lin1_b)   [4096, 4096]
  gemm_wmma<<<dim3(DFF_ / 64, ROWS / 64), 128, 0, stream>>>(
      hb, wb_l1, l1_b, nullptr, nullptr, ff1, ROWS, DFF_, D_, 1);
  // x2 = x1 + ff1 @ lin2_w^T + lin2_b  -> d_out
  gemm_wmma<<<dim3(D_ / 64, ROWS / 64), 128, 0, stream>>>(
      ff1, wb_l2, l2_b, x1, x2_out, nullptr, ROWS, D_, DFF_, 0);
}
